// BatchGAT_91268055040440
// MI455X (gfx1250) — matrix operand order repaired and checked by reading, not compiled, not run on hardware
//
#include <hip/hip_runtime.h>
#include <hip/hip_bf16.h>

typedef __attribute__((ext_vector_type(16))) _Float16 v16h;
typedef __attribute__((ext_vector_type(8)))  float    v8f;

#define NN 2048   // nodes per graph
#define FD 64     // f_in == f_out
// bh index = b*8 + h, 32 total

// ---------------------------------------------------------------------------
// Kernel 1: hp = h @ W[h]  (per b,h, 32-row block), tanh -> s,d dot products.
// Writes hpT (f16, [bh][o][n] transposed for WMMA-B contiguous loads),
// s, d, E=exp(d), F=exp(0.2 d).
// ---------------------------------------------------------------------------
__global__ __launch_bounds__(256) void gat_hprime_kernel(
    const float* __restrict__ h, const float* __restrict__ w,
    const float* __restrict__ a_src, const float* __restrict__ a_dst,
    _Float16* __restrict__ hpT, float* __restrict__ sArr,
    float* __restrict__ dArr, float* __restrict__ eArr, float* __restrict__ fArr)
{
    __shared__ float wl[FD * FD];   // 16 KB
    __shared__ float hl[32 * FD];   // 8 KB
    __shared__ float sAcc[32];
    __shared__ float dAcc[32];

    const int tid = threadIdx.x;
    const int bh  = blockIdx.x;
    const int b   = bh >> 3;
    const int hh  = bh & 7;
    const int r0  = blockIdx.y * 32;

    for (int i = tid; i < FD * FD; i += 256) wl[i] = w[hh * FD * FD + i];
    for (int i = tid; i < 32 * FD; i += 256)
        hl[i] = h[((size_t)b * NN + r0) * FD + i];
    if (tid < 32) { sAcc[tid] = 0.f; dAcc[tid] = 0.f; }
    __syncthreads();

    const int o  = tid & 63;
    const int rg = tid >> 6;              // 0..3 -> 8 rows each
    const float asv = a_src[hh * FD + o];
    const float adv = a_dst[hh * FD + o];

    for (int rr = 0; rr < 8; ++rr) {
        const int row = rg * 8 + rr;
        float acc = 0.f;
        #pragma unroll
        for (int f = 0; f < FD; ++f) acc += hl[row * FD + f] * wl[f * FD + o];
        const float t = tanhf(acc);
        atomicAdd(&sAcc[row], t * asv);   // ds_add_f32
        atomicAdd(&dAcc[row], t * adv);
        hpT[((size_t)bh * FD + o) * NN + r0 + row] = (_Float16)acc;
    }
    __syncthreads();

    if (tid < 32) {
        const size_t gi = (size_t)bh * NN + r0 + tid;
        const float dv = dAcc[tid];
        sArr[gi] = sAcc[tid];
        dArr[gi] = dv;
        eArr[gi] = __expf(dv);
        fArr[gi] = __expf(0.2f * dv);
    }
}

// ---------------------------------------------------------------------------
// Kernel 2: dmax[bh] = max_j d[bh][j]
// ---------------------------------------------------------------------------
__global__ __launch_bounds__(256) void gat_dmax_kernel(
    const float* __restrict__ dArr, float* __restrict__ dmax)
{
    __shared__ float red[256];
    const int tid = threadIdx.x, bh = blockIdx.x;
    float m = -3.4e38f;
    for (int i = tid; i < NN; i += 256) m = fmaxf(m, dArr[(size_t)bh * NN + i]);
    red[tid] = m; __syncthreads();
    #pragma unroll
    for (int s2 = 128; s2 > 0; s2 >>= 1) {
        if (tid < s2) red[tid] = fmaxf(red[tid], red[tid + s2]);
        __syncthreads();
    }
    if (tid == 0) dmax[bh] = red[0];
}

// ---------------------------------------------------------------------------
// Kernel 3: fused softmax(LR(s_i + d_j)) @ hp, flash-style.
// One wave per 16-row tile, 4 N-tiles + 1 ones-tile (row-sum Z) WMMAs / K-iter,
// K-loop unrolled x2 for load/WMMA overlap.
// ---------------------------------------------------------------------------
__global__ __launch_bounds__(256) void gat_attn_kernel(
    const _Float16* __restrict__ hpT, const float* __restrict__ sArr,
    const float* __restrict__ eArr, const float* __restrict__ fArr,
    const float* __restrict__ dmaxArr, const float* __restrict__ bias,
    float* __restrict__ out)
{
    const int bh   = blockIdx.x;
    const int wave = threadIdx.x >> 5;
    const int lane = threadIdx.x & 31;
    const int i0   = (blockIdx.y * 8 + wave) * 16;

    const int mrow = lane & 15;   // A-matrix row (M)
    const int hi   = lane >> 4;   // lane half
    const int kb   = hi * 16;     // A: this half covers K = kb + e
    const int kBb  = hi * 16;     // B: this half covers K = kBb + e
    const int nlan = lane & 15;   // B/C/D column (N)

    const float* sP = sArr + (size_t)bh * NN;
    const float* eP = eArr + (size_t)bh * NN;
    const float* fP = fArr + (size_t)bh * NN;
    const _Float16* hB = hpT + (size_t)bh * FD * NN;

    const float s_m   = sP[i0 + mrow];
    const float dmx   = dmaxArr[bh];
    const float x0    = s_m + dmx;
    const float c     = fmaxf(x0, 0.2f * x0);     // row max m_i = LR(s_i + dmax)
    const float alpha = __expf(s_m - c);
    const float beta  = __expf(0.2f * s_m - c);
    const float thr   = __expf(-s_m);             // E_j >= thr  <=>  s_i + d_j >= 0

    v8f acc0 = {}, acc1 = {}, acc2 = {}, acc3 = {}, accz = {};
    v16h ones;
    #pragma unroll
    for (int e = 0; e < 16; ++e) ones[e] = (_Float16)1.0f;

    #pragma unroll 2
    for (int j0 = 0; j0 < NN; j0 += 32) {
        const float4* ep = (const float4*)(eP + j0 + kb);
        const float4* fp = (const float4*)(fP + j0 + kb);
        float ev[16], fv[16];
        float4 t;
        t = ep[0]; ev[0]=t.x; ev[1]=t.y; ev[2]=t.z;  ev[3]=t.w;
        t = ep[1]; ev[4]=t.x; ev[5]=t.y; ev[6]=t.z;  ev[7]=t.w;
        t = ep[2]; ev[8]=t.x; ev[9]=t.y; ev[10]=t.z; ev[11]=t.w;
        t = ep[3]; ev[12]=t.x; ev[13]=t.y; ev[14]=t.z; ev[15]=t.w;
        t = fp[0]; fv[0]=t.x; fv[1]=t.y; fv[2]=t.z;  fv[3]=t.w;
        t = fp[1]; fv[4]=t.x; fv[5]=t.y; fv[6]=t.z;  fv[7]=t.w;
        t = fp[2]; fv[8]=t.x; fv[9]=t.y; fv[10]=t.z; fv[11]=t.w;
        t = fp[3]; fv[12]=t.x; fv[13]=t.y; fv[14]=t.z; fv[15]=t.w;

        v16h a;
        #pragma unroll
        for (int e = 0; e < 16; ++e) {
            const float wgt = (ev[e] >= thr) ? (alpha * ev[e]) : (beta * fv[e]);
            a[e] = (_Float16)wgt;
        }

        const v16h b0 = *(const v16h*)(hB + ((size_t)(nlan     )) * NN + j0 + kBb);
        const v16h b1 = *(const v16h*)(hB + ((size_t)(nlan + 16)) * NN + j0 + kBb);
        const v16h b2 = *(const v16h*)(hB + ((size_t)(nlan + 32)) * NN + j0 + kBb);
        const v16h b3 = *(const v16h*)(hB + ((size_t)(nlan + 48)) * NN + j0 + kBb);

        acc0 = __builtin_amdgcn_wmma_f32_16x16x32_f16(false, a, false, b0, (short)0, acc0, false, false);
        acc1 = __builtin_amdgcn_wmma_f32_16x16x32_f16(false, a, false, b1, (short)0, acc1, false, false);
        acc2 = __builtin_amdgcn_wmma_f32_16x16x32_f16(false, a, false, b2, (short)0, acc2, false, false);
        acc3 = __builtin_amdgcn_wmma_f32_16x16x32_f16(false, a, false, b3, (short)0, acc3, false, false);
        accz = __builtin_amdgcn_wmma_f32_16x16x32_f16(false, a, false, ones, (short)0, accz, false, false);
    }

    const float bias0 = bias[nlan];
    const float bias1 = bias[nlan + 16];
    const float bias2 = bias[nlan + 32];
    const float bias3 = bias[nlan + 48];
    float* outP = out + ((size_t)bh * NN + i0) * FD;
    #pragma unroll
    for (int r = 0; r < 8; ++r) {
        // softmax normalization: fast v_rcp_f32 is plenty (Z >= 1 by construction,
        // since the max-scoring term contributes exp(0) = 1).
        const float inv = __builtin_amdgcn_rcpf(accz[r]);
        const int gm = r + hi * 8;            // C/D layout: VGPR r holds M = r + 8*half
        float* op = outP + (size_t)gm * FD;
        op[nlan]      = acc0[r] * inv + bias0;
        op[nlan + 16] = acc1[r] * inv + bias1;
        op[nlan + 32] = acc2[r] * inv + bias2;
        op[nlan + 48] = acc3[r] * inv + bias3;
    }
}

// ---------------------------------------------------------------------------
extern "C" void kernel_launch(void* const* d_in, const int* in_sizes, int n_in,
                              void* d_out, int out_size, void* d_ws, size_t ws_size,
                              hipStream_t stream) {
    const float* h     = (const float*)d_in[0];
    // d_in[1] = adj (bool) -- unused by the reference computation
    const float* w     = (const float*)d_in[2];
    const float* a_src = (const float*)d_in[3];
    const float* a_dst = (const float*)d_in[4];
    const float* bias  = (const float*)d_in[5];
    float* out = (float*)d_out;

    char* ws = (char*)d_ws;
    const size_t HPT_BYTES = (size_t)32 * FD * NN * sizeof(_Float16); // 8 MiB
    const size_t VEC_BYTES = (size_t)32 * NN * sizeof(float);         // 256 KiB
    _Float16* hpT  = (_Float16*)ws;
    float*    sArr = (float*)(ws + HPT_BYTES);
    float*    dArr = (float*)(ws + HPT_BYTES + 1 * VEC_BYTES);
    float*    eArr = (float*)(ws + HPT_BYTES + 2 * VEC_BYTES);
    float*    fArr = (float*)(ws + HPT_BYTES + 3 * VEC_BYTES);
    float*    dmax = (float*)(ws + HPT_BYTES + 4 * VEC_BYTES);

    gat_hprime_kernel<<<dim3(32, NN / 32), 256, 0, stream>>>(
        h, w, a_src, a_dst, hpT, sArr, dArr, eArr, fArr);
    gat_dmax_kernel<<<dim3(32), 256, 0, stream>>>(dArr, dmax);
    gat_attn_kernel<<<dim3(32, NN / 128), 256, 0, stream>>>(
        hpT, sArr, eArr, fArr, dmax, bias, out);
}
